// AttentionAggregationNN_15625091023546
// MI455X (gfx1250) — matrix operand order repaired and checked by reading, not compiled, not run on hardware
//
#include <hip/hip_runtime.h>
#include <math.h>

// Problem constants (from reference)
#define EDIM   256
#define HDIM   8
#define DDIM   32
#define MAXM   128
#define LSTR   264   // LDS row stride in ushorts (256 + 8 pad; 528B, 16B-aligned)
#define SSTR   17    // scores LDS row stride in floats (bank-conflict-free)

typedef __attribute__((ext_vector_type(16))) __bf16 v16bf;
typedef __attribute__((ext_vector_type(8)))  float  v8f;

__device__ __forceinline__ unsigned short f2bf(float f) {
    unsigned int u = __float_as_uint(f);
    unsigned int r = (u + 0x7FFFu + ((u >> 16) & 1u)) >> 16;   // round-to-nearest-even
    return (unsigned short)r;
}

// ---------------------------------------------------------------------------
// Kernel 1: fold all weights. One block, 256 threads.
// Outputs:
//   bfrag: pre-swizzled WMMA B fragments, [8 kblk][32 lane][16 bf16]
//          column n<8  -> qk[h=n]   (query-folded K projection, scaled 1/sqrt(D))
//          column n>=8 -> wvo[h=n-8] (w_lin/w_out-folded V projection)
//   bscal: scalar bias b_all
// ---------------------------------------------------------------------------
__global__ __launch_bounds__(256)
void precompute_kernel(const float* __restrict__ query,
                       const float* __restrict__ w_in,  const float* __restrict__ b_in,
                       const float* __restrict__ w_out, const float* __restrict__ b_out,
                       const float* __restrict__ w_lin, const float* __restrict__ b_lin,
                       unsigned short* __restrict__ bfrag, float* __restrict__ bscal) {
    __shared__ float qsc[EDIM];       // scaled query projection q/sqrt(D), [h*32+d]
    __shared__ float wol[EDIM];       // w_lin @ w_out, [e]
    __shared__ float bcol[16 * EDIM]; // B matrix column-major [n][k]
    __shared__ float red[256];

    const int t = threadIdx.x;

    // q[o] = query . w_in[o,:] + b_in[o], scaled by 1/sqrt(32)
    float aq = b_in[t];
    for (int e = 0; e < EDIM; ++e) aq += query[e] * w_in[t * EDIM + e];
    qsc[t] = aq * 0.17677669529663687f;

    // wol[e] = sum_o w_lin[o] * w_out[o,e]
    float aw = 0.f;
    for (int o = 0; o < EDIM; ++o) aw += w_lin[o] * w_out[o * EDIM + t];
    wol[t] = aw;
    __syncthreads();

    // Fold K and V projections; thread t owns column e = t
    for (int h = 0; h < HDIM; ++h) {
        float ak = 0.f, av = 0.f;
        for (int d = 0; d < DDIM; ++d) {
            const int o = h * DDIM + d;
            ak += qsc[o] * w_in[(EDIM + o) * EDIM + t];        // wk rows
            av += w_in[(2 * EDIM + o) * EDIM + t] * wol[o];    // wv rows
        }
        bcol[h * EDIM + t]          = ak;
        bcol[(8 + h) * EDIM + t]    = av;
    }
    // bias folding: bv.wol + b_out.w_lin (+ b_lin added by thread 0)
    red[t] = b_in[2 * EDIM + t] * wol[t] + b_out[t] * w_lin[t];
    __syncthreads();

    // Emit pre-swizzled B fragments: contiguous k order (same as the A reads),
    // lane n(0..15)=col n holds K 0..15; lanes 16..31 hold K 16..31.
    {
        const int kblk = t >> 5, lane = t & 31;
        const int n = lane & 15;
        const int k0 = kblk * 32 + ((lane & 16) ? 16 : 0);
        #pragma unroll
        for (int j = 0; j < 16; ++j)
            bfrag[t * 16 + j] = f2bf(bcol[n * EDIM + k0 + j]);
    }
    for (int s = 128; s > 0; s >>= 1) {
        if (t < s) red[t] += red[t + s];
        __syncthreads();
    }
    if (t == 0) bscal[0] = red[0] + b_lin[0];
}

// ---------------------------------------------------------------------------
// Kernel 2: scatter row indices into per-group lists.
// ---------------------------------------------------------------------------
__global__ __launch_bounds__(256)
void scatter_kernel(const int* __restrict__ gid, int* __restrict__ cursor,
                    int* __restrict__ idx, int n) {
    const int i = blockIdx.x * blockDim.x + threadIdx.x;
    if (i >= n) return;
    const int g = gid[i];
    const int pos = atomicAdd(&cursor[g], 1);
    if (pos < MAXM) idx[g * MAXM + pos] = i;
}

// ---------------------------------------------------------------------------
// Kernel 3: per-group fused attention. One block (128 threads = 4 waves) per
// group. Gathers rows, converts to bf16 in LDS, runs a single
// (128x256)@(256x16) v_wmma_f32_16x16x32_bf16 GEMM (cols 0-7 = scores,
// cols 8-15 = t-matrix), masked softmax, weighted reduce to one scalar.
// ---------------------------------------------------------------------------
__global__ __launch_bounds__(128)
void attn_main_kernel(const float* __restrict__ tp, const int* __restrict__ idx,
                      const int* __restrict__ cursor,
                      const unsigned short* __restrict__ bfrag,
                      const float* __restrict__ bscal, float* __restrict__ out) {
    __shared__ __align__(16) unsigned short embS[MAXM * LSTR];  // 66 KB bf16 tile
    __shared__ float scT[MAXM * SSTR];                          // [m][16] scores|t
    __shared__ float hres[HDIM];

    const int g    = blockIdx.x;
    const int t    = threadIdx.x;
    const int lane = t & 31;
    const int wave = t >> 5;

    int size = cursor[g];
    if (size > MAXM) size = MAXM;

    // ---- gather rows -> bf16 LDS tile (2 rows per pass, float4 per thread)
    {
        const int c4 = t & 63;      // float4 slot within a row (64 per row)
        const int mo = t >> 6;      // 0/1: which of the two rows this pass
        for (int p = 0; p < 64; ++p) {
            const int m = p * 2 + mo;
            unsigned int p0 = 0u, p1 = 0u;
            if (m < size) {
                const int row = idx[g * MAXM + m];
                const float4 f = ((const float4*)tp)[row * 64 + c4];
                p0 = (unsigned int)f2bf(f.x) | ((unsigned int)f2bf(f.y) << 16);
                p1 = (unsigned int)f2bf(f.z) | ((unsigned int)f2bf(f.w) << 16);
            }
            *(uint2*)(&embS[m * LSTR + c4 * 4]) = make_uint2(p0, p1);
        }
    }
    __syncthreads();

    // ---- WMMA GEMM: D[128x16] = emb[128x256] @ B[256x16]
    // A layout (16-bit 16x32): lane 0-15 M=lane K{0..15}; lane 16-31
    // M=lane-16 K{16..31}  -> two 16B LDS reads per fragment.
    {
        const int mlane = lane & 15;
        const int koff  = (lane & 16) ? 16 : 0;
        for (int mb = wave; mb < 8; mb += 4) {          // 2 M-blocks per wave
            v8f c = {0.f, 0.f, 0.f, 0.f, 0.f, 0.f, 0.f, 0.f};
            const int mrow = mb * 16 + mlane;
            #pragma unroll
            for (int kb = 0; kb < 8; ++kb) {
                union { uint4 u[2]; v16bf v; } A, B;
                const uint4* ap =
                    (const uint4*)(&embS[mrow * LSTR + kb * 32 + koff]);
                A.u[0] = ap[0];          // K base .. +7
                A.u[1] = ap[1];          // K base+8 .. +15
                const uint4* bp = (const uint4*)(bfrag + (kb * 32 + lane) * 16);
                B.u[0] = bp[0];
                B.u[1] = bp[1];
                c = __builtin_amdgcn_wmma_f32_16x16x32_bf16(
                        false, A.v, false, B.v, (short)0, c, false, false);
            }
            // D layout: lane 0-15 -> N=lane, M=r; lane 16-31 -> N=lane-16, M=r+8
            const int mr = mb * 16 + ((lane & 16) ? 8 : 0);
            #pragma unroll
            for (int r = 0; r < 8; ++r)
                scT[(mr + r) * SSTR + mlane] = c[r];
        }
    }
    __syncthreads();

    // ---- masked softmax over m (cols 0-7) + weighted sum of t (cols 8-15)
    for (int h = wave * 2; h <= wave * 2 + 1; ++h) {
        float mx = -1e30f;
        for (int m = lane; m < size; m += 32)
            mx = fmaxf(mx, scT[m * SSTR + h]);
        for (int off = 16; off > 0; off >>= 1)
            mx = fmaxf(mx, __shfl_xor(mx, off, 32));
        float num = 0.f, den = 0.f;
        for (int m = lane; m < size; m += 32) {
            const float ev = __expf(scT[m * SSTR + h] - mx);
            den += ev;
            num += ev * scT[m * SSTR + 8 + h];
        }
        for (int off = 16; off > 0; off >>= 1) {
            num += __shfl_xor(num, off, 32);
            den += __shfl_xor(den, off, 32);
        }
        if (lane == 0) hres[h] = num / den;
    }
    __syncthreads();
    if (t == 0) {
        float s = bscal[0];
        #pragma unroll
        for (int h = 0; h < HDIM; ++h) s += hres[h];
        out[g] = s;
    }
}

// ---------------------------------------------------------------------------
extern "C" void kernel_launch(void* const* d_in, const int* in_sizes, int n_in,
                              void* d_out, int out_size, void* d_ws, size_t ws_size,
                              hipStream_t stream) {
    const float* tree_preds = (const float*)d_in[0];
    const int*   group_ids  = (const int*)d_in[1];
    const float* query      = (const float*)d_in[2];
    const float* w_in       = (const float*)d_in[3];
    const float* b_in       = (const float*)d_in[4];
    const float* w_out      = (const float*)d_in[5];
    const float* b_out      = (const float*)d_in[6];
    const float* w_lin      = (const float*)d_in[7];
    const float* b_lin      = (const float*)d_in[8];
    float*       out        = (float*)d_out;

    const int N = in_sizes[1];     // number of rows
    const int G = out_size;        // number of groups

    // workspace layout (256B-aligned sections)
    char* w = (char*)d_ws;
    size_t o0 = 0;                                         // cursor[G]
    size_t o1 = (o0 + (size_t)G * 4 + 255) & ~(size_t)255; // idx[G*MAXM]
    size_t o2 = (o1 + (size_t)G * MAXM * 4 + 255) & ~(size_t)255; // bfrag
    size_t o3 = (o2 + 8 * 32 * 16 * 2 + 255) & ~(size_t)255;      // bscal
    int*            cursor = (int*)(w + o0);
    int*            idx    = (int*)(w + o1);
    unsigned short* bfrag  = (unsigned short*)(w + o2);
    float*          bscal  = (float*)(w + o3);

    hipMemsetAsync(cursor, 0, (size_t)G * 4, stream);

    precompute_kernel<<<1, 256, 0, stream>>>(query, w_in, b_in, w_out, b_out,
                                             w_lin, b_lin, bfrag, bscal);
    scatter_kernel<<<(N + 255) / 256, 256, 0, stream>>>(group_ids, cursor, idx, N);
    attn_main_kernel<<<G, 128, 0, stream>>>(tree_preds, idx, cursor, bfrag,
                                            bscal, out);
}
